// EdgeMLP_76819785056586
// MI455X (gfx1250) — compile-verified, matrix operand order checked by reading, not run on hardware
//
#include <hip/hip_runtime.h>
#include <hip/hip_bf16.h>

// ---------------------------------------------------------------------------
// EdgeMLP + pairwise cosine similarity with class mask, CDNA5 (gfx1250).
//
// Phase 1: per-row MLP (3->64->32), row-normalized, emitted as f16 + int label.
// Phase 2: 8192x8192 cosine matrix via v_wmma_f32_16x16x32_f16 (K=32 exactly
//          one WMMA per 16x16 tile), label mask fused, NT streaming stores.
// ---------------------------------------------------------------------------

typedef _Float16 v16h __attribute__((ext_vector_type(16)));
typedef _Float16 v8h  __attribute__((ext_vector_type(8)));
typedef float    v8f  __attribute__((ext_vector_type(8)));

#define N_ROWS 8192
#define FDIM   32
#define HDIM   64

// ---------------------------------------------------------------------------
// Phase 1: MLP + normalize.  One thread per row; weights staged in LDS.
// ---------------------------------------------------------------------------
__global__ void edge_mlp_norm_kernel(const float* __restrict__ edges,  // [N,4]
                                     const float* __restrict__ W1,     // [3,64]
                                     const float* __restrict__ b1,     // [64]
                                     const float* __restrict__ W2,     // [64,32]
                                     const float* __restrict__ b2,     // [32]
                                     _Float16*    __restrict__ fOut,   // [N,32] f16
                                     int*         __restrict__ lblOut) // [N]
{
    __shared__ float sW1[3 * HDIM];
    __shared__ float sb1[HDIM];
    __shared__ float sW2[HDIM * FDIM];
    __shared__ float sb2[FDIM];

    const int tid = threadIdx.x;
    for (int i = tid; i < 3 * HDIM;    i += blockDim.x) sW1[i] = W1[i];
    for (int i = tid; i < HDIM;        i += blockDim.x) sb1[i] = b1[i];
    for (int i = tid; i < HDIM * FDIM; i += blockDim.x) sW2[i] = W2[i];
    for (int i = tid; i < FDIM;        i += blockDim.x) sb2[i] = b2[i];
    __syncthreads();

    const int row = blockIdx.x * blockDim.x + tid;
    if (row >= N_ROWS) return;

    const float x0 = edges[row * 4 + 0];
    const float x1 = edges[row * 4 + 1];
    const float x2 = edges[row * 4 + 2];
    const int   lb = (int)edges[row * 4 + 3];

    float f[FDIM];
#pragma unroll
    for (int j = 0; j < FDIM; ++j) f[j] = sb2[j];

    // h_k = relu(x . W1[:,k] + b1[k]);  f += h_k * W2[k,:]
    for (int k = 0; k < HDIM; ++k) {
        float h = sb1[k] + x0 * sW1[0 * HDIM + k]
                          + x1 * sW1[1 * HDIM + k]
                          + x2 * sW1[2 * HDIM + k];
        h = fmaxf(h, 0.0f);
#pragma unroll
        for (int j = 0; j < FDIM; ++j) f[j] += h * sW2[k * FDIM + j];
    }

    float nsq = 0.0f;
#pragma unroll
    for (int j = 0; j < FDIM; ++j) nsq += f[j] * f[j];
    // denom = max(n1*n2, 1e-8): fold eps per-side so the big GEMM emits cosine
    const float rn = rsqrtf(fmaxf(nsq, 1e-16f));

#pragma unroll
    for (int j = 0; j < FDIM; ++j)
        fOut[row * FDIM + j] = (_Float16)(f[j] * rn);
    lblOut[row] = lb;
}

// ---------------------------------------------------------------------------
// Phase 2: pairwise cosine + mask.  One wave per 16x16 tile x 8 column tiles.
//
// f16 16x32 A-fragment layout (ISA 7.12.2): lane L<16 -> row L, k in
// {0..7, 16..23}; lane L>=16 -> row L-16, k in {8..15, 24..31}.  B (=f2^T)
// columns are f2 rows, with the identical per-lane gather pattern.
// ---------------------------------------------------------------------------
#define COLS_PER_WAVE 8
#define WAVES_PER_BLOCK 4

__global__ void pairwise_cosine_wmma_kernel(const _Float16* __restrict__ fA,
                                            const _Float16* __restrict__ fB,
                                            const int*      __restrict__ lA,
                                            const int*      __restrict__ lB,
                                            float*          __restrict__ out)
{
    const int lane = threadIdx.x & 31;
    const int wave = threadIdx.x >> 5;
    const int gw   = blockIdx.x * WAVES_PER_BLOCK + wave;

    const int colGroups = (N_ROWS / 16) / COLS_PER_WAVE;   // 64
    const int rowTile   = gw / colGroups;                  // 0..511
    const int colGroup  = gw % colGroups;                  // 0..63

    const int half  = lane >> 4;      // 0 or 1
    const int l15   = lane & 15;
    const int kBase = half * 8;       // k chunk start: 0 or 8

    // ---- A fragment: row = rowTile*16 + l15, k in {kBase..+7, kBase+16..+23}
    const int aRow = rowTile * 16 + l15;
    const _Float16* aPtr = fA + aRow * FDIM + kBase;
    v8h aLo = *(const v8h*)(aPtr);        // 16B -> global_load_b128
    v8h aHi = *(const v8h*)(aPtr + 16);
    v16h a;
#pragma unroll
    for (int i = 0; i < 8; ++i) { a[i] = aLo[i]; a[i + 8] = aHi[i]; }

    // Row labels for the 8 D registers (uniform per half-wave, hoisted).
    int rlbl[8];
#pragma unroll
    for (int r = 0; r < 8; ++r)
        rlbl[r] = lA[rowTile * 16 + r + half * 8];

    for (int ct = 0; ct < COLS_PER_WAVE; ++ct) {
        const int colTile = colGroup * COLS_PER_WAVE + ct;

        // ---- B fragment: column n of B = row n of fB, same gather as A
        const int bRow = colTile * 16 + l15;
        const _Float16* bPtr = fB + bRow * FDIM + kBase;
        v8h bLo = *(const v8h*)(bPtr);
        v8h bHi = *(const v8h*)(bPtr + 16);
        v16h b;
#pragma unroll
        for (int i = 0; i < 8; ++i) { b[i] = bLo[i]; b[i + 8] = bHi[i]; }

        const int clbl = lB[bRow];   // column label for N = l15 (all D regs)

        v8f c = {};
        // D = A x B + 0  ->  v_wmma_f32_16x16x32_f16
        v8f d = __builtin_amdgcn_wmma_f32_16x16x32_f16(
            /*neg_a=*/false, a, /*neg_b=*/false, b,
            /*c_mod=*/(short)0, c, /*reuse_a=*/false, /*reuse_b=*/false);

        // D layout: VGPR r -> (M = r + half*8, N = l15).  Mask + NT store.
        const int colIdx = colTile * 16 + l15;
#pragma unroll
        for (int r = 0; r < 8; ++r) {
            const int   rowIdx = rowTile * 16 + r + half * 8;
            const float v      = (rlbl[r] == clbl) ? d[r] : 0.0f;
            __builtin_nontemporal_store(v, out + (size_t)rowIdx * N_ROWS + colIdx);
        }
    }
}

// ---------------------------------------------------------------------------
// Launch
// ---------------------------------------------------------------------------
extern "C" void kernel_launch(void* const* d_in, const int* in_sizes, int n_in,
                              void* d_out, int out_size, void* d_ws, size_t ws_size,
                              hipStream_t stream)
{
    const float* edges1 = (const float*)d_in[0];  // [8192,4]
    const float* edges2 = (const float*)d_in[1];  // [8192,4]
    const float* W1     = (const float*)d_in[2];  // [3,64]
    const float* b1     = (const float*)d_in[3];  // [64]
    const float* W2     = (const float*)d_in[4];  // [64,32]
    const float* b2     = (const float*)d_in[5];  // [32]
    float*       out    = (float*)d_out;          // [8192,8192]

    // Workspace carve-out: fA, fB (f16), lA, lB (int)
    _Float16* fA = (_Float16*)d_ws;
    _Float16* fB = fA + (size_t)N_ROWS * FDIM;
    int*      lA = (int*)(fB + (size_t)N_ROWS * FDIM);
    int*      lB = lA + N_ROWS;

    // Phase 1: MLP + normalize for both edge sets
    {
        dim3 block(256);
        dim3 grid(N_ROWS / 256);
        edge_mlp_norm_kernel<<<grid, block, 0, stream>>>(edges1, W1, b1, W2, b2, fA, lA);
        edge_mlp_norm_kernel<<<grid, block, 0, stream>>>(edges2, W1, b1, W2, b2, fB, lB);
    }

    // Phase 2: 512 row tiles x 64 col groups = 32768 waves, 4 waves/block
    {
        const int totalWaves = (N_ROWS / 16) * ((N_ROWS / 16) / COLS_PER_WAVE);
        dim3 block(WAVES_PER_BLOCK * 32);
        dim3 grid(totalWaves / WAVES_PER_BLOCK);
        pairwise_cosine_wmma_kernel<<<grid, block, 0, stream>>>(fA, fB, lA, lB, out);
    }
}
